// SwinTransformerBlock_79345225826870
// MI455X (gfx1250) — matrix operand order repaired and checked by reading, not compiled, not run on hardware
//
#include <hip/hip_runtime.h>

// ---------------------------------------------------------------------------
// Types for CDNA5 WMMA (wave32): v_wmma_f32_16x16x32_bf16
// ---------------------------------------------------------------------------
typedef __attribute__((ext_vector_type(16))) __bf16 v16bf;
typedef __attribute__((ext_vector_type(8)))  float  v8f;
typedef __attribute__((ext_vector_type(4)))  unsigned int v4u;
typedef __attribute__((ext_vector_type(8)))  int    v8i;
typedef __attribute__((ext_vector_type(4)))  int    v4i;

union ABfrag {
    v16bf v;
    unsigned int   u[8];
    unsigned short s[16];
};

static __device__ inline __bf16 f2bf(float f) {
    union { float f; unsigned int u; } c; c.f = f;
    unsigned int r = c.u + 0x7FFFu + ((c.u >> 16) & 1u);   // round-to-nearest-even
    union { unsigned short s; __bf16 b; } o; o.s = (unsigned short)(r >> 16);
    return o.b;
}
static __device__ inline float bf2f(__bf16 b) {
    union { unsigned short s; __bf16 b; } i; i.b = b;
    union { unsigned int u; float f; } o; o.u = ((unsigned int)i.s) << 16;
    return o.f;
}

static __device__ inline v8f wmma_bf16(v16bf a, v16bf b, v8f c) {
    return __builtin_amdgcn_wmma_f32_16x16x32_bf16(false, a, false, b,
                                                   (short)0, c, false, false);
}

// A-fragment (16x32 bf16, M x K): lane holds row M=lane%16; element pair r,s:
//   K = (r>=4?16:0) + 2*(r%4) + s + (lane/16)*8    (ISA 7.12.2)
static __device__ inline v16bf load_afrag(const __bf16* __restrict__ rowptr,
                                          int half, int k0) {
    ABfrag a;
#pragma unroll
    for (int r = 0; r < 8; ++r) {
        int kk = k0 + ((r & 4) << 2) + ((r & 3) << 1) + (half << 3);
        a.u[r] = *(const unsigned int*)(rowptr + kk);
    }
    return a.v;
}

// B-fragment (32x16 bf16, K x N): lane holds N = lane%16; element s:
//   K = (s>=8?16:0) + (s%8) + (lane/16)*8, row stride ld (strided gather).
static __device__ inline v16bf load_bfrag(const __bf16* __restrict__ p, int ld, int half) {
    ABfrag b;
#pragma unroll
    for (int s = 0; s < 16; ++s) {
        int kk = ((s & 8) << 1) + (half << 3) + (s & 7);
        b.s[s] = *(const unsigned short*)(p + kk * ld);
    }
    return b.v;
}

// ---------------------------------------------------------------------------
// Tensor Data Mover: DMA a 32(K) x 128(N) bf16 tile of W into LDS.
// D# built per CDNA5 ISA ch.8: Group0 {count, lds_addr, global_addr, type=2},
// Group1 {data_size=2B, tensor_dim0=N, tensor_dim1=rows, tile=128x32,
//         tensor_dim0_stride=N}. Remaining groups zero (2-D tensor).
// This toolchain exposes the 6-arg builtin form.
// ---------------------------------------------------------------------------
static __device__ inline void tdm_load_b_tile(const __bf16* gsrc,
                                              unsigned lds_addr,
                                              int N, int rowsLeft) {
    unsigned long long ga = (unsigned long long)(size_t)gsrc;
    v4u g0;
    g0[0] = 1u;                                        // count=1, no gather
    g0[1] = lds_addr;                                  // LDS byte address
    g0[2] = (unsigned)(ga & 0xFFFFFFFFull);            // global_addr[31:0]
    g0[3] = (unsigned)((ga >> 32) & 0x01FFFFFFull)     // global_addr[56:32]
          | 0x80000000u;                               // type=2 ("image")
    v8i g1;
    g1[0] = 0x00010000;                                // data_size=1 (2 bytes)
    g1[1] = (int)(((unsigned)N & 0xFFFFu) << 16);      // tensor_dim0[15:0]
    g1[2] = (int)((((unsigned)N >> 16) & 0xFFFFu)      // tensor_dim0[31:16]
          | (((unsigned)rowsLeft & 0xFFFFu) << 16));   // tensor_dim1[15:0]
    g1[3] = (int)((((unsigned)rowsLeft >> 16) & 0xFFFFu)
          | (128u << 16));                             // tile_dim0 = 128
    g1[4] = 32;                                        // tile_dim1=32, tile_dim2=0
    g1[5] = N;                                         // tensor_dim0_stride[31:0]
    g1[6] = 0;                                         // stride0[47:32], stride1 lo
    g1[7] = 0;
    v4i z4 = {0, 0, 0, 0};
    v8i z8 = {0, 0, 0, 0, 0, 0, 0, 0};
    __builtin_amdgcn_tensor_load_to_lds(g0, g1, z4, z4, z8, 0);
}

// ---------------------------------------------------------------------------
// fp32 -> bf16 conversion (weights)
// ---------------------------------------------------------------------------
__global__ __launch_bounds__(256)
void cvt_f32_to_bf16(const float* __restrict__ src, __bf16* __restrict__ dst, int n) {
    for (int i = blockIdx.x * 256 + threadIdx.x; i < n; i += gridDim.x * 256)
        dst[i] = f2bf(src[i]);
}

// ---------------------------------------------------------------------------
// LayerNorm over D=1024, one block (256 threads) per token, bf16 output
// ---------------------------------------------------------------------------
__global__ __launch_bounds__(256)
void layernorm_to_bf16(const float* __restrict__ x, const float* __restrict__ g,
                       const float* __restrict__ b, __bf16* __restrict__ out) {
    __shared__ float red[256];
    const int row = blockIdx.x;
    const float* xr = x + (size_t)row * 1024;
    float4 v = ((const float4*)xr)[threadIdx.x];

    red[threadIdx.x] = v.x + v.y + v.z + v.w;
    __syncthreads();
    for (int off = 128; off > 0; off >>= 1) {
        if (threadIdx.x < off) red[threadIdx.x] += red[threadIdx.x + off];
        __syncthreads();
    }
    float mu = red[0] * (1.f / 1024.f);
    __syncthreads();

    float d0 = v.x - mu, d1 = v.y - mu, d2 = v.z - mu, d3 = v.w - mu;
    red[threadIdx.x] = d0 * d0 + d1 * d1 + d2 * d2 + d3 * d3;
    __syncthreads();
    for (int off = 128; off > 0; off >>= 1) {
        if (threadIdx.x < off) red[threadIdx.x] += red[threadIdx.x + off];
        __syncthreads();
    }
    float rstd = rsqrtf(red[0] * (1.f / 1024.f) + 1e-5f);

    float4 gg = ((const float4*)g)[threadIdx.x];
    float4 bb = ((const float4*)b)[threadIdx.x];
    __bf16* orow = out + (size_t)row * 1024;
    int c = threadIdx.x * 4;
    orow[c + 0] = f2bf(d0 * rstd * gg.x + bb.x);
    orow[c + 1] = f2bf(d1 * rstd * gg.y + bb.y);
    orow[c + 2] = f2bf(d2 * rstd * gg.z + bb.z);
    orow[c + 3] = f2bf(d3 * rstd * gg.w + bb.w);
}

// ---------------------------------------------------------------------------
// bf16 WMMA GEMM with TDM-staged B tiles:
//   block (8 waves) computes 128(M) x 128(N); wave = 32(M) x 64(N).
//   Wave 0 DMAs the 32x128 W-slice into LDS (double buffered, TENSORcnt),
//   all waves read B fragments from LDS. A fragments from global (L2-hot).
// residMode: 0 none, 1 fp32, 2 bf16.  outMode: 0 bf16, 1 fp32.
// ---------------------------------------------------------------------------
__global__ __launch_bounds__(256)
void gemm_bf16(const __bf16* __restrict__ A, const __bf16* __restrict__ W,
               const float* __restrict__ bias,
               const void* __restrict__ resid, int residMode,
               void* __restrict__ out, int outMode,
               int M, int N, int K) {
    __shared__ __align__(16) __bf16 Bt[2][32 * 128];

    const int wid  = threadIdx.x >> 5;
    const int lane = threadIdx.x & 31;
    const int half = lane >> 4;
    const int l16  = lane & 15;
    const int mquad = wid >> 1;          // 0..3 : which 32-row strip
    const int nhalf = wid & 1;           // 0..1 : which 64-col strip
    const int tilesN = N >> 7;
    const int bm = blockIdx.x / tilesN;
    const int bn = blockIdx.x % tilesN;
    const int rowBase = bm * 128 + mquad * 32;
    const int colBase = bn * 128;

    v8f acc[2][4];
#pragma unroll
    for (int s = 0; s < 2; ++s)
#pragma unroll
        for (int i = 0; i < 4; ++i)
#pragma unroll
            for (int j = 0; j < 8; ++j) acc[s][i][j] = 0.f;

    const __bf16* Arow0 = A + (size_t)(rowBase + l16) * K;
    const __bf16* Arow1 = A + (size_t)(rowBase + 16 + l16) * K;
    const int kB = l16 + (half << 4);
    const int nSteps = K >> 5;

    if (wid == 0)
        tdm_load_b_tile(W + colBase, (unsigned)(size_t)&Bt[0][0], N, K);

    int p = 0;
    for (int i = 0; i < nSteps; ++i) {
        if (wid == 0) __builtin_amdgcn_s_wait_tensorcnt(0);
        __syncthreads();                 // tile i visible; buf p^1 free
        if (wid == 0 && i + 1 < nSteps)
            tdm_load_b_tile(W + (size_t)((i + 1) << 5) * N + colBase,
                            (unsigned)(size_t)&Bt[p ^ 1][0],
                            N, K - ((i + 1) << 5));
        const int k0 = i << 5;
        v16bf a0 = load_afrag(Arow0, half, k0);
        v16bf a1 = load_afrag(Arow1, half, k0);
        const __bf16* Bp = &Bt[p][nhalf * 64 + l16];
#pragma unroll
        for (int nt = 0; nt < 4; ++nt) {
            v16bf bf = load_bfrag(Bp + nt * 16, 128, half);
            acc[0][nt] = wmma_bf16(a0, bf, acc[0][nt]);
            acc[1][nt] = wmma_bf16(a1, bf, acc[1][nt]);
        }
        __syncthreads();                 // done reading buf p
        p ^= 1;
    }

#pragma unroll
    for (int s = 0; s < 2; ++s) {
#pragma unroll
        for (int nt = 0; nt < 4; ++nt) {
            const int n = colBase + nhalf * 64 + nt * 16 + l16;
            const float bv = bias[n];
#pragma unroll
            for (int r = 0; r < 8; ++r) {
                const int m = rowBase + s * 16 + r + (half << 3);
                const size_t idx = (size_t)m * N + n;
                float v = acc[s][nt][r] + bv;
                if (residMode == 1)      v += ((const float*)resid)[idx];
                else if (residMode == 2) v += bf2f(((const __bf16*)resid)[idx]);
                if (outMode == 0) ((__bf16*)out)[idx] = f2bf(v);
                else              ((float*)out)[idx]  = v;
            }
        }
    }
}

// ---------------------------------------------------------------------------
// Windowed MHSA: one block (128 threads, 4 waves) per (window, head).
// Partition / merge / cyclic roll folded into p0[]; shift mask via groups.
// ---------------------------------------------------------------------------
__global__ __launch_bounds__(128)
void win_attn(const __bf16* __restrict__ qkv, __bf16* __restrict__ outp,
              int shifted) {
    __shared__ __align__(16) __bf16 Kt[64 * 64];   // transposed K: [dh][tok]
    __shared__ __align__(16) float  S [64 * 64];   // scores
    __shared__ __align__(16) __bf16 P [64 * 64];   // softmax probs
    __shared__ int p0s[64];
    __shared__ int grp[64];

    const int winId = blockIdx.x;
    const int h     = blockIdx.y;
    const int batch = winId >> 6;
    const int win   = winId & 63;
    const int wy = win >> 3, wx = win & 7;
    const int tid = threadIdx.x;

    if (tid < 64) {
        int ry = tid >> 3, rx = tid & 7;
        int r = wy * 8 + ry, c = wx * 8 + rx;
        int gr = r, gc = c;
        if (shifted) { gr = (r + 4) & 63; gc = (c + 4) & 63; }
        p0s[tid] = batch * 4096 + gr * 64 + gc;
        int rr = (r < 56) ? 0 : (r < 60 ? 1 : 2);
        int rc = (c < 56) ? 0 : (c < 60 ? 1 : 2);
        grp[tid] = rr * 3 + rc;
    }
    __syncthreads();

    {   // stage K transposed into LDS
        int tok = tid >> 1, hh = tid & 1;
        const __bf16* kr = qkv + (size_t)p0s[tok] * 3072 + 1024 + h * 64 + hh * 32;
#pragma unroll
        for (int j = 0; j < 32; ++j) Kt[(hh * 32 + j) * 64 + tok] = kr[j];
    }
    __syncthreads();

    const int wid = tid >> 5, lane = tid & 31;
    const int half = lane >> 4, l16 = lane & 15;
    const int m0 = wid * 16;
    const int kB = l16 + (half << 4);

    v8f acc[4];
#pragma unroll
    for (int i = 0; i < 4; ++i)
#pragma unroll
        for (int j = 0; j < 8; ++j) acc[i][j] = 0.f;

    const __bf16* qrow = qkv + (size_t)p0s[m0 + l16] * 3072 + h * 64;
    for (int k0 = 0; k0 < 64; k0 += 32) {
        v16bf af = load_afrag(qrow, half, k0);
        const __bf16* Bp = &Kt[k0 * 64 + l16];
#pragma unroll
        for (int nt = 0; nt < 4; ++nt) {
            v16bf bf = load_bfrag(Bp + nt * 16, 64, half);
            acc[nt] = wmma_bf16(af, bf, acc[nt]);
        }
    }
#pragma unroll
    for (int nt = 0; nt < 4; ++nt) {
        int n = nt * 16 + l16;
#pragma unroll
        for (int r = 0; r < 8; ++r) {
            int m = m0 + r + (half << 3);
            float v = acc[nt][r] * 0.125f;
            if (shifted && (grp[m] != grp[n])) v -= 1e9f;
            S[m * 64 + n] = v;
        }
    }
    __syncthreads();

    if (tid < 64) {
        float mx = -3.4e38f;
        for (int j = 0; j < 64; ++j) mx = fmaxf(mx, S[tid * 64 + j]);
        float sum = 0.f;
        for (int j = 0; j < 64; ++j) {
            float e = __expf(S[tid * 64 + j] - mx);
            S[tid * 64 + j] = e;
            sum += e;
        }
        float inv = 1.f / sum;
        for (int j = 0; j < 64; ++j) P[tid * 64 + j] = f2bf(S[tid * 64 + j] * inv);
    }
    __syncthreads();

    v8f oacc[4];
#pragma unroll
    for (int i = 0; i < 4; ++i)
#pragma unroll
        for (int j = 0; j < 8; ++j) oacc[i][j] = 0.f;

    const __bf16* prow = &P[(m0 + l16) * 64];
    for (int k0 = 0; k0 < 64; k0 += 32) {
        v16bf af = load_afrag(prow, half, k0);
        const __bf16* vbase = qkv + 2048 + h * 64 + l16;
#pragma unroll
        for (int nt = 0; nt < 4; ++nt) {
            ABfrag bv;
#pragma unroll
            for (int s = 0; s < 16; ++s) {
                int kk = k0 + ((s & 8) << 1) + (half << 3) + (s & 7);
                bv.s[s] = *(const unsigned short*)(vbase + (size_t)p0s[kk] * 3072 + nt * 16);
            }
            oacc[nt] = wmma_bf16(af, bv.v, oacc[nt]);
        }
    }
#pragma unroll
    for (int nt = 0; nt < 4; ++nt) {
        int n = nt * 16 + l16;
#pragma unroll
        for (int r = 0; r < 8; ++r) {
            int m = m0 + r + (half << 3);
            outp[(size_t)p0s[m] * 1024 + h * 64 + n] = f2bf(oacc[nt][r]);
        }
    }
}

// ---------------------------------------------------------------------------
// Host launcher
// ---------------------------------------------------------------------------
extern "C" void kernel_launch(void* const* d_in, const int* in_sizes, int n_in,
                              void* d_out, int out_size, void* d_ws, size_t ws_size,
                              hipStream_t stream) {
    (void)in_sizes; (void)n_in; (void)out_size; (void)ws_size;
    const int M = 16384;

    const float* x      = (const float*)d_in[0];
    const float* ln1_g  = (const float*)d_in[1];
    const float* ln1_b  = (const float*)d_in[2];
    const float* ln2_g  = (const float*)d_in[3];
    const float* ln2_b  = (const float*)d_in[4];
    const float* ln3_g  = (const float*)d_in[5];
    const float* ln3_b  = (const float*)d_in[6];
    const float* ln4_g  = (const float*)d_in[7];
    const float* ln4_b  = (const float*)d_in[8];
    const float* m1a_w  = (const float*)d_in[9];
    const float* m1a_b  = (const float*)d_in[10];
    const float* m1b_w  = (const float*)d_in[11];
    const float* m1b_b  = (const float*)d_in[12];
    const float* m2a_w  = (const float*)d_in[13];
    const float* m2a_b  = (const float*)d_in[14];
    const float* m2b_w  = (const float*)d_in[15];
    const float* m2b_b  = (const float*)d_in[16];
    const float* a1_wqkv = (const float*)d_in[17];
    const float* a1_bqkv = (const float*)d_in[18];
    const float* a1_wo   = (const float*)d_in[19];
    const float* a1_bo   = (const float*)d_in[20];
    const float* a2_wqkv = (const float*)d_in[21];
    const float* a2_bqkv = (const float*)d_in[22];
    const float* a2_wo   = (const float*)d_in[23];
    const float* a2_bo   = (const float*)d_in[24];

    char* ws = (char*)d_ws;
    __bf16* bfw   = (__bf16*)ws;
    __bf16* Wqkv1 = bfw + 0;
    __bf16* Wo1   = bfw + 3145728;
    __bf16* M1a   = bfw + 4194304;
    __bf16* M1b   = bfw + 8388608;
    __bf16* Wqkv2 = bfw + 12582912;
    __bf16* Wo2   = bfw + 15728640;
    __bf16* M2a   = bfw + 16777216;
    __bf16* M2b   = bfw + 20971520;
    size_t off = 25165824ull * 2;
    __bf16* a_bf    = (__bf16*)(ws + off); off += 33554432ull;   // [M,1024]
    __bf16* qkv_bf  = (__bf16*)(ws + off); off += 100663296ull;  // [M,3072]
    __bf16* attn_bf = (__bf16*)(ws + off); off += 33554432ull;   // [M,1024]
    __bf16* h_bf    = (__bf16*)(ws + off); off += 134217728ull;  // [M,4096]
    float*  wbuf    = (float*) (ws + off); off += 67108864ull;   // w / sw
    float*  out1f   = (float*) (ws + off); off += 67108864ull;   // out1

    auto cvt = [&](const float* s, __bf16* d, int n) {
        cvt_f32_to_bf16<<<2048, 256, 0, stream>>>(s, d, n);
    };
    cvt(a1_wqkv, Wqkv1, 3145728);
    cvt(a1_wo,   Wo1,   1048576);
    cvt(m1a_w,   M1a,   4194304);
    cvt(m1b_w,   M1b,   4194304);
    cvt(a2_wqkv, Wqkv2, 3145728);
    cvt(a2_wo,   Wo2,   1048576);
    cvt(m2a_w,   M2a,   4194304);
    cvt(m2b_w,   M2b,   4194304);

    // grid = (M/128) * (N/128) blocks of 256 threads
    // ---------- layer 1 : W-MSA + MLP ----------
    layernorm_to_bf16<<<M, 256, 0, stream>>>(x, ln1_g, ln1_b, a_bf);
    gemm_bf16<<<128 * 24, 256, 0, stream>>>(a_bf, Wqkv1, a1_bqkv,
                                            nullptr, 0, qkv_bf, 0, M, 3072, 1024);
    win_attn<<<dim3(256, 16), 128, 0, stream>>>(qkv_bf, attn_bf, 0);
    gemm_bf16<<<128 * 8, 256, 0, stream>>>(attn_bf, Wo1, a1_bo,
                                           a_bf, 2, wbuf, 1, M, 1024, 1024);
    layernorm_to_bf16<<<M, 256, 0, stream>>>(wbuf, ln2_g, ln2_b, a_bf);
    gemm_bf16<<<128 * 32, 256, 0, stream>>>(a_bf, M1a, m1a_b,
                                            nullptr, 0, h_bf, 0, M, 4096, 1024);
    gemm_bf16<<<128 * 8, 256, 0, stream>>>(h_bf, M1b, m1b_b,
                                           wbuf, 1, out1f, 1, M, 1024, 4096);

    // ---------- layer 2 : SW-MSA + MLP ----------
    layernorm_to_bf16<<<M, 256, 0, stream>>>(out1f, ln3_g, ln3_b, a_bf);
    gemm_bf16<<<128 * 24, 256, 0, stream>>>(a_bf, Wqkv2, a2_bqkv,
                                            nullptr, 0, qkv_bf, 0, M, 3072, 1024);
    win_attn<<<dim3(256, 16), 128, 0, stream>>>(qkv_bf, attn_bf, 1);
    gemm_bf16<<<128 * 8, 256, 0, stream>>>(attn_bf, Wo2, a2_bo,
                                           out1f, 1, wbuf, 1, M, 1024, 1024);
    layernorm_to_bf16<<<M, 256, 0, stream>>>(wbuf, ln4_g, ln4_b, a_bf);
    gemm_bf16<<<128 * 32, 256, 0, stream>>>(a_bf, M2a, m2a_b,
                                            nullptr, 0, h_bf, 0, M, 4096, 1024);
    gemm_bf16<<<128 * 8, 256, 0, stream>>>(h_bf, M2b, m2b_b,
                                           wbuf, 1, (float*)d_out, 1, M, 1024, 4096);
}
